// Attention_81484119540519
// MI455X (gfx1250) — matrix operand order repaired and checked by reading, not compiled, not run on hardware
//
#include <hip/hip_runtime.h>

typedef __attribute__((ext_vector_type(16))) __bf16 v16bf;
typedef __attribute__((ext_vector_type(8)))  float  v8f;

union Frag16 { v16bf bf; unsigned u[8]; };
union Pk2 { __bf16 h[2]; unsigned u; };

__device__ __forceinline__ unsigned pack_bf2(float a, float b) {
  Pk2 p; p.h[0] = (__bf16)a; p.h[1] = (__bf16)b; return p.u;
}
__device__ __forceinline__ unsigned pack_bf2h(__bf16 a, __bf16 b) {
  Pk2 p; p.h[0] = a; p.h[1] = b; return p.u;
}

// A-matrix fragment: 16(M) x 32(K) bf16. row_base points at this lane's row
// (caller adds (lane&15)*stride). ISA layout: lanes 0-15 K-pairs {0,1..6,7} in
// v0-3, {16..23} in v4-7; lanes 16-31 shifted by +8 in K.
__device__ __forceinline__ v16bf load_matA_k32(const __bf16* row_base, int lane) {
  const int kbase = (lane >> 4) << 3;
  Frag16 f;
#pragma unroll
  for (int v = 0; v < 8; ++v) {
    const int k0 = (v < 4) ? (kbase + 2 * v) : (16 + kbase + 2 * (v - 4));
    f.u[v] = *reinterpret_cast<const unsigned*>(row_base + k0);
  }
  return f.bf;
}

// B-matrix fragment: 32(K) x 16(N) bf16 from LDS laid out [n][k].
// col_base points at this lane's column row (caller adds (lane&15)*stride).
// Lanes 0-15 hold K=0..7 in v0-3 and K=16..23 in v4-7; lanes 16-31 shifted by +8 in K.
__device__ __forceinline__ v16bf load_matB_k32(const __bf16* col_base, int lane) {
  const int kbase = (lane >> 4) << 3;
  Frag16 f;
#pragma unroll
  for (int v = 0; v < 8; ++v)
    f.u[v] = *reinterpret_cast<const unsigned*>(col_base + ((v < 4) ? (kbase + 2 * v) : (16 + kbase + 2 * (v - 4))));
  return f.bf;
}

__device__ __forceinline__ void lds_wave_fence() {
  asm volatile("s_wait_dscnt 0" ::: "memory");
}

#define NH 8
#define HW 1024
#define CDIM 512
#define SCALE 0.17677669529663687f
#define L2E 1.44269504088896340f

// ---------------------------------------------------------------------------
// Kernel 1: QKV projection. qkv[o,n] = BN(sum_c w[o,c] * x[b,c,n]).
// 128x128 tile per WG, 8 waves (wave32), K staged 32 at a time in LDS (bf16).
// Routes output rows into q(bf16)/k(bf16)/v(f32) head-major buffers.
// ---------------------------------------------------------------------------
__global__ __launch_bounds__(256) void qkv_gemm(
    const float* __restrict__ x, const float* __restrict__ w,
    const float* __restrict__ g, const float* __restrict__ bb,
    __bf16* __restrict__ qb, __bf16* __restrict__ kb, float* __restrict__ vf) {
  __shared__ __bf16 aS[128 * 34];  // [o_local][k], stride 34 (bank-friendly)
  __shared__ __bf16 bS[128 * 34];  // [n_local][k]
  const int tid = threadIdx.x, lane = tid & 31, wave = tid >> 5;
  const int b = blockIdx.z;
  const int o0 = blockIdx.x * 128, n0 = blockIdx.y * 128;
  const size_t xb = (size_t)b * CDIM * HW;
  const int wr = (wave >> 1) * 32;  // wave row block (2x16 rows)
  const int wc = (wave & 1) * 64;   // wave col block (4x16 cols)

  v8f zero = {};
  v8f acc[2][4];
#pragma unroll
  for (int i = 0; i < 2; ++i)
#pragma unroll
    for (int j = 0; j < 4; ++j) acc[i][j] = zero;

  for (int k0 = 0; k0 < CDIM; k0 += 32) {
    // Stage A: w_qkv rows (fp32 -> bf16), vectorized float4.
#pragma unroll
    for (int it = 0; it < 4; ++it) {
      int e = tid + 256 * it;          // 1024 float4's
      int row = e >> 3, kq = (e & 7) * 4;
      const float4 t =
          *reinterpret_cast<const float4*>(w + (size_t)(o0 + row) * CDIM + k0 + kq);
      unsigned* d = reinterpret_cast<unsigned*>(&aS[row * 34 + kq]);
      d[0] = pack_bf2(t.x, t.y);
      d[1] = pack_bf2(t.z, t.w);
    }
    // Stage B transposed: x[b][c][n] -> bS[n][k] so B-fragments load as dwords.
#pragma unroll
    for (int j = 0; j < 8; ++j) {
      int p = tid + 256 * j;           // 2048 k-pairs
      int n = p & 127, kp = p >> 7;
      int c = k0 + kp * 2;
      float f0 = x[xb + (size_t)c * HW + n0 + n];
      float f1 = x[xb + (size_t)(c + 1) * HW + n0 + n];
      *reinterpret_cast<unsigned*>(&bS[n * 34 + kp * 2]) = pack_bf2(f0, f1);
    }
    __syncthreads();

    v16bf afr[2], bfr[4];
#pragma unroll
    for (int rb = 0; rb < 2; ++rb)
      afr[rb] = load_matA_k32(&aS[(wr + rb * 16 + (lane & 15)) * 34], lane);
#pragma unroll
    for (int cb = 0; cb < 4; ++cb)
      bfr[cb] = load_matB_k32(&bS[(wc + cb * 16 + (lane & 15)) * 34], lane);
#pragma unroll
    for (int rb = 0; rb < 2; ++rb)
#pragma unroll
      for (int cb = 0; cb < 4; ++cb)
        acc[rb][cb] = __builtin_amdgcn_wmma_f32_16x16x32_bf16(
            false, afr[rb], false, bfr[cb], (short)0, acc[rb][cb], false, false);
    __syncthreads();
  }

  // Epilogue: BN + route rows to q/k/v. C-layout: elem v -> row +((lane>>4)*8+v),
  // col = lane&15.
#pragma unroll
  for (int rb = 0; rb < 2; ++rb)
#pragma unroll
    for (int cb = 0; cb < 4; ++cb) {
      const int ob = o0 + wr + rb * 16 + ((lane >> 4) << 3);
      const int n = n0 + wc + cb * 16 + (lane & 15);
#pragma unroll
      for (int v = 0; v < 8; ++v) {
        const int o = ob + v;
        const float val = acc[rb][cb][v] * g[o] + bb[o];
        const int h = o >> 7, r = o & 127;
        const size_t hb = (size_t)b * NH + h;
        if (r < 32)
          qb[(hb * 32 + r) * HW + n] = (__bf16)val;
        else if (r < 64)
          kb[(hb * 32 + (r - 32)) * HW + n] = (__bf16)val;
        else
          vf[(hb * 64 + (r - 64)) * HW + n] = val;
      }
    }
}

// ---------------------------------------------------------------------------
// Kernel 2: flash attention. One WG per (b,h,64-query tile); 4 waves, each
// owns 16 queries. Loops 32-key chunks: S = Qt*K (WMMA, K=d_k=32), online
// softmax (shfl_xor reductions within 16-lane halves), P staged per-wave in
// LDS as the B operand for O += V*Pt (4 WMMAs over d_v=64).
// ---------------------------------------------------------------------------
__global__ __launch_bounds__(128) void flash_attn(
    const __bf16* __restrict__ qb, const __bf16* __restrict__ kb,
    const float* __restrict__ vf, float* __restrict__ att) {
  __shared__ __bf16 kS[32 * 34];       // [m][d]  (B-operand layout for scores)
  __shared__ __bf16 vS[64 * 34];       // [d][m]  (A-operand layout for O)
  __shared__ __bf16 pS[4][16 * 34];    // per-wave [n][m] (B-operand for O)
  __shared__ float cS[4][16];          // per-wave column correction broadcast
  __shared__ float sS[4][16];          // per-wave final row sums

  const int tid = threadIdx.x, lane = tid & 31, wave = tid >> 5;
  const int b = blockIdx.z, h = blockIdx.y;
  const int nq0 = blockIdx.x * 64 + wave * 16;
  const int nl = lane & 15;
  const int kbase = (lane >> 4) << 3;
  const size_t hq = ((size_t)b * NH + h) * 32 * HW;
  const size_t hv = ((size_t)b * NH + h) * 64 * HW;

  // Q fragment (persistent): A elem (M=n_local, K=d) = q[d][nq0+nl].
  Frag16 qa;
  {
    const __bf16* qp = qb + hq + nq0 + nl;
#pragma unroll
    for (int v = 0; v < 8; ++v) {
      const int d0 = (v < 4) ? (kbase + 2 * v) : (16 + kbase + 2 * (v - 4));
      qa.u[v] = pack_bf2h(qp[(size_t)d0 * HW], qp[(size_t)(d0 + 1) * HW]);
    }
  }

  float mx[8], sm[8];
#pragma unroll
  for (int v = 0; v < 8; ++v) { mx[v] = -1.0e30f; sm[v] = 0.0f; }
  v8f zero = {};
  v8f oacc[4] = {zero, zero, zero, zero};

  for (int m0 = 0; m0 < HW; m0 += 32) {
    // Stage K chunk: kS[m][d] (pack 2 consecutive d per dword).
#pragma unroll
    for (int j = 0; j < 4; ++j) {
      int p = tid + 128 * j;           // 512 k-pairs
      int m = p & 31, d = (p >> 5) * 2;
      __bf16 lo = kb[hq + (size_t)d * HW + m0 + m];
      __bf16 hi = kb[hq + (size_t)(d + 1) * HW + m0 + m];
      *reinterpret_cast<unsigned*>(&kS[m * 34 + d]) = pack_bf2h(lo, hi);
    }
    // Stage V chunk (f32 -> bf16): vS[d][m].
#pragma unroll
    for (int j = 0; j < 4; ++j) {
      int p = tid + 128 * j;           // 512 float4's
      int d = p >> 3, mq = (p & 7) * 4;
      const float4 t =
          *reinterpret_cast<const float4*>(vf + hv + (size_t)d * HW + m0 + mq);
      unsigned* dst = reinterpret_cast<unsigned*>(&vS[d * 34 + mq]);
      dst[0] = pack_bf2(t.x, t.y);
      dst[1] = pack_bf2(t.z, t.w);
    }
    __syncthreads();

    // Scores: two 16x16 tiles (keys m0..+15, m0+16..+31), K = d_k = 32.
    v16bf kb0 = load_matB_k32(&kS[nl * 34], lane);
    v16bf kb1 = load_matB_k32(&kS[(16 + nl) * 34], lane);
    v8f s0 = __builtin_amdgcn_wmma_f32_16x16x32_bf16(
        false, qa.bf, false, kb0, (short)0, zero, false, false);
    v8f s1 = __builtin_amdgcn_wmma_f32_16x16x32_bf16(
        false, qa.bf, false, kb1, (short)0, zero, false, false);

    // Online softmax. C layout: elem v -> row n = (lane>>4)*8+v, col = lane&15.
    float corrA[8];
#pragma unroll
    for (int v = 0; v < 8; ++v) {
      float a0 = s0[v] * SCALE, a1 = s1[v] * SCALE;
      float rm = fmaxf(a0, a1);
      rm = fmaxf(rm, __shfl_xor(rm, 1));
      rm = fmaxf(rm, __shfl_xor(rm, 2));
      rm = fmaxf(rm, __shfl_xor(rm, 4));
      rm = fmaxf(rm, __shfl_xor(rm, 8));
      const float nm = fmaxf(mx[v], rm);
      const float corr = exp2f((mx[v] - nm) * L2E);
      const float p0 = exp2f((a0 - nm) * L2E);
      const float p1 = exp2f((a1 - nm) * L2E);
      float rs = p0 + p1;
      rs += __shfl_xor(rs, 1);
      rs += __shfl_xor(rs, 2);
      rs += __shfl_xor(rs, 4);
      rs += __shfl_xor(rs, 8);
      sm[v] = sm[v] * corr + rs;
      mx[v] = nm;
      corrA[v] = corr;
      const int n = kbase + v;
      pS[wave][n * 34 + nl] = (__bf16)p0;
      pS[wave][n * 34 + 16 + nl] = (__bf16)p1;
    }
    if (nl == 0) {
#pragma unroll
      for (int v = 0; v < 8; ++v) cS[wave][kbase + v] = corrA[v];
    }
    lds_wave_fence();

    // Rescale accumulated O by this column's correction, then O += V * Pt.
    const float mycorr = cS[wave][nl];
#pragma unroll
    for (int blk = 0; blk < 4; ++blk)
#pragma unroll
      for (int v = 0; v < 8; ++v) oacc[blk][v] *= mycorr;

    v16bf pb = load_matB_k32(&pS[wave][nl * 34], lane);
#pragma unroll
    for (int blk = 0; blk < 4; ++blk) {
      v16bf va = load_matA_k32(&vS[(blk * 16 + nl) * 34], lane);
      oacc[blk] = __builtin_amdgcn_wmma_f32_16x16x32_bf16(
          false, va, false, pb, (short)0, oacc[blk], false, false);
    }
    __syncthreads();
  }

  // Normalize by row sums and store attended[b][c=h*64+d][n] (f32).
  if (nl == 0) {
#pragma unroll
    for (int v = 0; v < 8; ++v) sS[wave][kbase + v] = sm[v];
  }
  lds_wave_fence();
  const float inv = 1.0f / sS[wave][nl];
  const size_t ab = ((size_t)b * CDIM + h * 64) * HW;
#pragma unroll
  for (int blk = 0; blk < 4; ++blk) {
    const int db = blk * 16 + kbase;
#pragma unroll
    for (int v = 0; v < 8; ++v)
      att[ab + (size_t)(db + v) * HW + nq0 + nl] = oacc[blk][v] * inv;
  }
}

// ---------------------------------------------------------------------------
// Kernel 3: depthwise 3x3 conv on v_spatial + BN + add attended -> enhanced
// (stored bf16 to feed the final GEMM directly).
// ---------------------------------------------------------------------------
__global__ __launch_bounds__(256) void pos_conv(
    const float* __restrict__ vf, const float* __restrict__ wpos,
    const float* __restrict__ gp, const float* __restrict__ bp,
    const float* __restrict__ att, __bf16* __restrict__ enh) {
  const int gid = blockIdx.x * 256 + threadIdx.x;
  if (gid >= 16 * CDIM * HW) return;
  const int n = gid & 1023;
  const int c = (gid >> 10) & 511;
  const int b = gid >> 19;
  const int xx = n & 31, yy = n >> 5;
  const float* vch = vf + ((size_t)b * CDIM + c) * HW;
  const float* wk = wpos + c * 9;
  float sum = 0.0f;
#pragma unroll
  for (int dy = -1; dy <= 1; ++dy) {
    const int y2 = yy + dy;
    if ((unsigned)y2 >= 32u) continue;
#pragma unroll
    for (int dx = -1; dx <= 1; ++dx) {
      const int x2 = xx + dx;
      if ((unsigned)x2 >= 32u) continue;
      sum += vch[y2 * 32 + x2] * wk[(dy + 1) * 3 + (dx + 1)];
    }
  }
  enh[gid] = (__bf16)(att[gid] + sum * gp[c] + bp[c]);
}

// ---------------------------------------------------------------------------
// Kernel 4: output projection. out[o,n] = BN(sum_c w_out[o,c]*enhanced[c,n]).
// Same WMMA-GEMM structure as kernel 1; B operand is already bf16.
// ---------------------------------------------------------------------------
__global__ __launch_bounds__(256) void out_gemm(
    const __bf16* __restrict__ enh, const float* __restrict__ w,
    const float* __restrict__ g, const float* __restrict__ bb,
    float* __restrict__ out) {
  __shared__ __bf16 aS[128 * 34];
  __shared__ __bf16 bS[128 * 34];
  const int tid = threadIdx.x, lane = tid & 31, wave = tid >> 5;
  const int b = blockIdx.z;
  const int o0 = blockIdx.x * 128, n0 = blockIdx.y * 128;
  const size_t eb = (size_t)b * CDIM * HW;
  const int wr = (wave >> 1) * 32;
  const int wc = (wave & 1) * 64;

  v8f zero = {};
  v8f acc[2][4];
#pragma unroll
  for (int i = 0; i < 2; ++i)
#pragma unroll
    for (int j = 0; j < 4; ++j) acc[i][j] = zero;

  for (int k0 = 0; k0 < CDIM; k0 += 32) {
#pragma unroll
    for (int it = 0; it < 4; ++it) {
      int e = tid + 256 * it;
      int row = e >> 3, kq = (e & 7) * 4;
      const float4 t =
          *reinterpret_cast<const float4*>(w + (size_t)(o0 + row) * CDIM + k0 + kq);
      unsigned* d = reinterpret_cast<unsigned*>(&aS[row * 34 + kq]);
      d[0] = pack_bf2(t.x, t.y);
      d[1] = pack_bf2(t.z, t.w);
    }
#pragma unroll
    for (int j = 0; j < 8; ++j) {
      int p = tid + 256 * j;
      int n = p & 127, kp = p >> 7;
      int c = k0 + kp * 2;
      __bf16 f0 = enh[eb + (size_t)c * HW + n0 + n];
      __bf16 f1 = enh[eb + (size_t)(c + 1) * HW + n0 + n];
      *reinterpret_cast<unsigned*>(&bS[n * 34 + kp * 2]) = pack_bf2h(f0, f1);
    }
    __syncthreads();

    v16bf afr[2], bfr[4];
#pragma unroll
    for (int rb = 0; rb < 2; ++rb)
      afr[rb] = load_matA_k32(&aS[(wr + rb * 16 + (lane & 15)) * 34], lane);
#pragma unroll
    for (int cb = 0; cb < 4; ++cb)
      bfr[cb] = load_matB_k32(&bS[(wc + cb * 16 + (lane & 15)) * 34], lane);
#pragma unroll
    for (int rb = 0; rb < 2; ++rb)
#pragma unroll
      for (int cb = 0; cb < 4; ++cb)
        acc[rb][cb] = __builtin_amdgcn_wmma_f32_16x16x32_bf16(
            false, afr[rb], false, bfr[cb], (short)0, acc[rb][cb], false, false);
    __syncthreads();
  }

#pragma unroll
  for (int rb = 0; rb < 2; ++rb)
#pragma unroll
    for (int cb = 0; cb < 4; ++cb) {
      const int ob = o0 + wr + rb * 16 + ((lane >> 4) << 3);
      const int n = n0 + wc + cb * 16 + (lane & 15);
#pragma unroll
      for (int v = 0; v < 8; ++v) {
        const int o = ob + v;
        out[((size_t)b * CDIM + o) * HW + n] = acc[rb][cb][v] * g[o] + bb[o];
      }
    }
}

// ---------------------------------------------------------------------------
extern "C" void kernel_launch(void* const* d_in, const int* in_sizes, int n_in,
                              void* d_out, int out_size, void* d_ws, size_t ws_size,
                              hipStream_t stream) {
  (void)in_sizes; (void)n_in; (void)out_size; (void)ws_size;
  const float* x     = (const float*)d_in[0];
  const float* w_qkv = (const float*)d_in[1];
  const float* g_qkv = (const float*)d_in[2];
  const float* b_qkv = (const float*)d_in[3];
  const float* w_pos = (const float*)d_in[4];
  const float* g_pos = (const float*)d_in[5];
  const float* b_pos = (const float*)d_in[6];
  const float* w_out = (const float*)d_in[7];
  const float* g_out = (const float*)d_in[8];
  const float* b_out = (const float*)d_in[9];

  // Workspace layout (bytes):
  //   [0, 8Mi)    q  bf16  [B,NH,32,HW]
  //   [8, 16Mi)   k  bf16  [B,NH,32,HW]
  //   [16, 48Mi)  v  f32   [B,C,HW]   (== v_spatial, c = h*64+d)
  //   [48, 80Mi)  attended f32 [B,C,HW]
  //   [80, 96Mi)  enhanced bf16 [B,C,HW]
  char* ws = (char*)d_ws;
  __bf16* qb  = (__bf16*)(ws);
  __bf16* kb  = (__bf16*)(ws + ((size_t)8 << 20));
  float*  vf  = (float*)(ws + ((size_t)16 << 20));
  float*  att = (float*)(ws + ((size_t)48 << 20));
  __bf16* enh = (__bf16*)(ws + ((size_t)80 << 20));

  qkv_gemm<<<dim3(8, 8, 16), 256, 0, stream>>>(x, w_qkv, g_qkv, b_qkv, qb, kb, vf);
  flash_attn<<<dim3(16, 8, 16), 128, 0, stream>>>(qb, kb, vf, att);
  pos_conv<<<dim3((16 * 512 * 1024) / 256), 256, 0, stream>>>(vf, w_pos, g_pos,
                                                              b_pos, att, enh);
  out_gemm<<<dim3(4, 8, 16), 256, 0, stream>>>(enh, w_out, g_out, b_out,
                                               (float*)d_out);
}
